// OneFormerDinatLayer_17884243821360
// MI455X (gfx1250) — matrix operand order repaired and checked by reading, not compiled, not run on hardware
//
#include <hip/hip_runtime.h>
#include <hip/hip_bf16.h>
#include <math.h>

typedef __attribute__((ext_vector_type(16))) _Float16 v16h;
typedef __attribute__((ext_vector_type(8)))  _Float16 v8h;
typedef __attribute__((ext_vector_type(8)))  float    v8f;

#define HH 64
#define WW 64
#define NPIX 4096
#define CC 256
#define HEADS 8
#define HD 32
#define KWIN 7
#define DIL 2
#define FF 1024
#define EPSV 1e-5f
#define SCALEV 0.17677669529663687f  // 1/sqrt(32)

// ---------------------------------------------------------------------------
// Weight convert + transpose: w[k][n] (f32, KxN) -> wt[n][k] (f16, NxK)
// ---------------------------------------------------------------------------
__global__ void convert_transpose_kernel(const float* __restrict__ w,
                                         _Float16* __restrict__ wt,
                                         int Kd, int Nd, float scale) {
    int total = Kd * Nd;
    for (int idx = blockIdx.x * blockDim.x + threadIdx.x; idx < total;
         idx += gridDim.x * blockDim.x) {
        int k = idx / Nd;
        int n = idx - k * Nd;
        wt[(size_t)n * Kd + k] = (_Float16)(w[idx] * scale);
    }
}

// Concatenate q/k/v biases (q-bias pre-scaled)
__global__ void concat_bias_kernel(const float* __restrict__ bq,
                                   const float* __restrict__ bk,
                                   const float* __restrict__ bv,
                                   float* __restrict__ out) {
    int t = blockIdx.x * blockDim.x + threadIdx.x;
    if (t < CC)            out[t] = bq[t] * SCALEV;
    else if (t < 2 * CC)   out[t] = bk[t - CC];
    else if (t < 3 * CC)   out[t] = bv[t - 2 * CC];
}

// ---------------------------------------------------------------------------
// LayerNorm over C=256, one wave (32 lanes) per token, f16 output
// ---------------------------------------------------------------------------
__global__ void layernorm_f16_kernel(const float* __restrict__ x,
                                     const float* __restrict__ g,
                                     const float* __restrict__ b,
                                     _Float16* __restrict__ out) {
    const int wid  = threadIdx.x >> 5;
    const int lane = threadIdx.x & 31;
    const int pix  = blockIdx.x * 8 + wid;
    if (pix >= NPIX) return;
    const float* xr = x + (size_t)pix * CC;

    float vals[8];
    float s = 0.f, sq = 0.f;
#pragma unroll
    for (int i = 0; i < 8; ++i) {
        float v = xr[lane + i * 32];
        vals[i] = v;
        s += v; sq += v * v;
    }
#pragma unroll
    for (int off = 16; off; off >>= 1) {
        s  += __shfl_xor(s,  off, 32);
        sq += __shfl_xor(sq, off, 32);
    }
    const float mean = s * (1.0f / CC);
    const float var  = sq * (1.0f / CC) - mean * mean;
    const float rstd = rsqrtf(var + EPSV);

    _Float16* orow = out + (size_t)pix * CC;
#pragma unroll
    for (int i = 0; i < 8; ++i) {
        int c = lane + i * 32;
        orow[c] = (_Float16)((vals[i] - mean) * rstd * g[c] + b[c]);
    }
}

// ---------------------------------------------------------------------------
// Generic WMMA GEMM: out[M,N] = A[M,K](f16) @ Bt[N,K]^T(f16) + bias
// Block = 256 threads = 8 waves (2 M-tiles x 4 N-tiles) -> 32x64 block tile.
// ---------------------------------------------------------------------------
union AFrag { v16h v; v8h h[2]; };

template <bool DO_GELU, bool HAS_RES, bool HALF_OUT>
__global__ void gemm_wmma_kernel(const _Float16* __restrict__ A,
                                 const _Float16* __restrict__ Bt,
                                 const float* __restrict__ bias,
                                 const float* __restrict__ res,
                                 float* __restrict__ outF,
                                 _Float16* __restrict__ outH,
                                 int M, int N, int Kd) {
    const int tid  = threadIdx.x;
    const int lane = tid & 31;
    const int wid  = tid >> 5;
    const int wm   = wid >> 2;   // 0..1
    const int wn   = wid & 3;    // 0..3
    const int half = lane >> 4;  // 0 or 1
    const int mrow = lane & 15;

    const int tileM = blockIdx.x * 32 + wm * 16;
    const int tileN = blockIdx.y * 64 + wn * 16;

    const _Float16* Arow = A  + (size_t)(tileM + mrow) * Kd;
    const _Float16* Brow = Bt + (size_t)(tileN + mrow) * Kd;

    v8f acc = {};
    // A 16-bit 16x32 layout: lanes 0-15 hold K 0..15; lanes 16-31 hold K 16..31 (contig.)
    // B 16-bit 32x16 layout: lanes 0-15 hold K 0..15; lanes 16-31 hold K 16..31 (contig.)
    for (int k0 = 0; k0 < Kd; k0 += 32) {
        AFrag af;
        af.v = *(const v16h*)(Arow + k0 + half * 16);
        v16h bf = *(const v16h*)(Brow + k0 + half * 16);
        acc = __builtin_amdgcn_wmma_f32_16x16x32_f16(
            false, af.v, false, bf, (short)0, acc, false, false);
    }

    const int   col  = mrow;                 // N offset within tile
    const float bval = bias[tileN + col];
#pragma unroll
    for (int r = 0; r < 8; ++r) {
        int row = tileM + r + 8 * half;      // C/D layout: VGPR r -> M=r (lanes<16) / M=8+r
        size_t idx = (size_t)row * N + tileN + col;
        float v = acc[r] + bval;
        if (DO_GELU) v = 0.5f * v * (1.0f + erff(v * 0.70710678118654752f));
        if (HAS_RES) v += res[idx];
        if (HALF_OUT) outH[idx] = (_Float16)v;
        else          outF[idx] = v;
    }
}

// ---------------------------------------------------------------------------
// Dilated neighborhood attention. One block per pixel; wave h handles head h,
// lane = channel within head. qkv layout: [pix][0:256]=q(scaled) [256:512]=k
// [512:768]=v. ctx output f16 [NPIX x 256].
// ---------------------------------------------------------------------------
__device__ __forceinline__ void axis_idx(int i, int* kidx, int* bidx) {
    const int g = i / DIL, rem = i - g * DIL;
    const int lg = (HH - rem + DIL - 1) / DIL;
    int start = g - (KWIN / 2);
    if (start < 0) start = 0;
    if (start > lg - KWIN) start = lg - KWIN;
#pragma unroll
    for (int a = 0; a < KWIN; ++a) {
        kidx[a] = (start + a) * DIL + rem;
        bidx[a] = start + a - g + (KWIN - 1);
    }
}

__global__ void natten_kernel(const float* __restrict__ qkv,
                              const float* __restrict__ rpb,
                              _Float16* __restrict__ ctx) {
    __shared__ float sc[HEADS][64];

    const int pix  = blockIdx.x;
    const int i    = pix >> 6;
    const int j    = pix & 63;
    const int h    = threadIdx.x >> 5;
    const int lane = threadIdx.x & 31;

    int ki_[KWIN], bi_[KWIN], kj_[KWIN], bj_[KWIN];
    axis_idx(i, ki_, bi_);
    axis_idx(j, kj_, bj_);

    const float qv = qkv[(size_t)pix * 768 + h * HD + lane];

    // scores = q . k_nb + rpb
    for (int p = 0; p < KWIN * KWIN; ++p) {
        const int a = p / KWIN, bb = p - a * KWIN;
        const int nb = ki_[a] * WW + kj_[bb];
        float d = qv * qkv[(size_t)nb * 768 + 256 + h * HD + lane];
#pragma unroll
        for (int off = 16; off; off >>= 1) d += __shfl_xor(d, off, 32);
        if (lane == 0)
            sc[h][p] = d + rpb[h * (13 * 13) + bi_[a] * 13 + bj_[bb]];
    }
    __syncthreads();

    // softmax over 49 (per head wave; lane covers p and p+32)
    float v0 = (lane      < 49) ? sc[h][lane]      : -3.4e38f;
    float v1 = (lane + 32 < 49) ? sc[h][lane + 32] : -3.4e38f;
    float m = fmaxf(v0, v1);
#pragma unroll
    for (int off = 16; off; off >>= 1) m = fmaxf(m, __shfl_xor(m, off, 32));
    float e0 = (lane      < 49) ? __expf(v0 - m) : 0.f;
    float e1 = (lane + 32 < 49) ? __expf(v1 - m) : 0.f;
    float s = e0 + e1;
#pragma unroll
    for (int off = 16; off; off >>= 1) s += __shfl_xor(s, off, 32);
    const float inv = 1.0f / s;
    if (lane < 49)      sc[h][lane]      = e0 * inv;
    if (lane + 32 < 49) sc[h][lane + 32] = e1 * inv;
    __syncthreads();

    // ctx = probs . v_nb
    float acc = 0.f;
    for (int p = 0; p < KWIN * KWIN; ++p) {
        const int a = p / KWIN, bb = p - a * KWIN;
        const int nb = ki_[a] * WW + kj_[bb];
        acc += sc[h][p] * qkv[(size_t)nb * 768 + 512 + h * HD + lane];
    }
    ctx[(size_t)pix * CC + h * HD + lane] = (_Float16)acc;
}

// ---------------------------------------------------------------------------
extern "C" void kernel_launch(void* const* d_in, const int* in_sizes, int n_in,
                              void* d_out, int out_size, void* d_ws, size_t ws_size,
                              hipStream_t stream) {
    const float* x     = (const float*)d_in[0];
    const float* ln1_g = (const float*)d_in[1];
    const float* ln1_b = (const float*)d_in[2];
    const float* wq    = (const float*)d_in[3];
    const float* bq    = (const float*)d_in[4];
    const float* wk    = (const float*)d_in[5];
    const float* bk    = (const float*)d_in[6];
    const float* wv    = (const float*)d_in[7];
    const float* bv    = (const float*)d_in[8];
    const float* rpb   = (const float*)d_in[9];
    const float* wo    = (const float*)d_in[10];
    const float* bo    = (const float*)d_in[11];
    const float* ln2_g = (const float*)d_in[12];
    const float* ln2_b = (const float*)d_in[13];
    const float* w1    = (const float*)d_in[14];
    const float* b1    = (const float*)d_in[15];
    const float* w2    = (const float*)d_in[16];
    const float* b2    = (const float*)d_in[17];
    float* out = (float*)d_out;

    // workspace carve (256B aligned)
    char* p = (char*)d_ws;
    auto alloc = [&](size_t bytes) -> char* {
        char* r = p;
        p += (bytes + 255) & ~(size_t)255;
        return r;
    };
    _Float16* wqkvT  = (_Float16*)alloc((size_t)768 * CC * 2);   // [768][256]
    float*    qkvB   = (float*)   alloc(768 * 4);
    _Float16* woT    = (_Float16*)alloc((size_t)CC * CC * 2);    // [256][256]
    _Float16* w1T    = (_Float16*)alloc((size_t)FF * CC * 2);    // [1024][256]
    _Float16* w2T    = (_Float16*)alloc((size_t)CC * FF * 2);    // [256][1024]
    _Float16* hs_h   = (_Float16*)alloc((size_t)NPIX * CC * 2);
    float*    qkv_f  = (float*)   alloc((size_t)NPIX * 768 * 4);
    _Float16* ctx_h  = (_Float16*)alloc((size_t)NPIX * CC * 2);
    float*    hs2_f  = (float*)   alloc((size_t)NPIX * CC * 4);
    _Float16* y2_h   = (_Float16*)alloc((size_t)NPIX * CC * 2);
    _Float16* ffn_h  = (_Float16*)alloc((size_t)NPIX * FF * 2);
    (void)ws_size; (void)n_in; (void)in_sizes; (void)out_size;

    // 1) weights -> f16 transposed (fold SCALE into wq/bq)
    convert_transpose_kernel<<<256, 256, 0, stream>>>(wq, wqkvT,            CC, CC, SCALEV);
    convert_transpose_kernel<<<256, 256, 0, stream>>>(wk, wqkvT + CC * CC,  CC, CC, 1.0f);
    convert_transpose_kernel<<<256, 256, 0, stream>>>(wv, wqkvT + 2*CC*CC,  CC, CC, 1.0f);
    convert_transpose_kernel<<<256, 256, 0, stream>>>(wo, woT,              CC, CC, 1.0f);
    convert_transpose_kernel<<<512, 256, 0, stream>>>(w1, w1T,              CC, FF, 1.0f);
    convert_transpose_kernel<<<512, 256, 0, stream>>>(w2, w2T,              FF, CC, 1.0f);
    concat_bias_kernel<<<3, 256, 0, stream>>>(bq, bk, bv, qkvB);

    // 2) LN1 -> hs (f16)
    layernorm_f16_kernel<<<NPIX / 8, 256, 0, stream>>>(x, ln1_g, ln1_b, hs_h);

    // 3) QKV = hs @ [wq|wk|wv] + bias   (M=4096, N=768, K=256)
    gemm_wmma_kernel<false, false, false><<<dim3(NPIX / 32, 768 / 64), 256, 0, stream>>>(
        hs_h, wqkvT, qkvB, nullptr, qkv_f, nullptr, NPIX, 768, CC);

    // 4) dilated neighborhood attention -> ctx (f16)
    natten_kernel<<<NPIX, 256, 0, stream>>>(qkv_f, rpb, ctx_h);

    // 5) hs2 = x + ctx @ wo + bo       (M=4096, N=256, K=256)
    gemm_wmma_kernel<false, true, false><<<dim3(NPIX / 32, CC / 64), 256, 0, stream>>>(
        ctx_h, woT, bo, x, hs2_f, nullptr, NPIX, CC, CC);

    // 6) LN2 -> y2 (f16)
    layernorm_f16_kernel<<<NPIX / 8, 256, 0, stream>>>(hs2_f, ln2_g, ln2_b, y2_h);

    // 7) ffn = gelu(y2 @ w1 + b1)      (M=4096, N=1024, K=256)
    gemm_wmma_kernel<true, false, true><<<dim3(NPIX / 32, FF / 64), 256, 0, stream>>>(
        y2_h, w1T, b1, nullptr, nullptr, ffn_h, NPIX, FF, CC);

    // 8) out = hs2 + ffn @ w2 + b2     (M=4096, N=256, K=1024)
    gemm_wmma_kernel<false, true, false><<<dim3(NPIX / 32, CC / 64), 256, 0, stream>>>(
        ffn_h, w2T, b2, hs2_f, out, nullptr, NPIX, CC, FF);
}
